// DCM_23484881174769
// MI455X (gfx1250) — matrix operand order repaired and checked by reading, not compiled, not run on hardware
//
#include <hip/hip_runtime.h>
#include <math.h>

// ---------------------------------------------------------------------------
// Problem constants (from the reference)
// ---------------------------------------------------------------------------
#define NWAY   5
#define NQRY   300
#define CDIM   640
#define LA     125      // a patches
#define LB     25       // b patches
#define LA_P   128      // padded
#define LB_P   32       // padded
#define DMODEL 128      // inner
#define EDIM   384      // 3*inner

typedef __attribute__((ext_vector_type(16))) _Float16 v16h;
typedef __attribute__((ext_vector_type(8)))  _Float16 v8h;
typedef __attribute__((ext_vector_type(8)))  float    v8f;

#define WMMA_F16(A, B, C) \
  __builtin_amdgcn_wmma_f32_16x16x32_f16(false, (A), false, (B), (short)0, (C), false, false)

// ---------------------------------------------------------------------------
// Fragment loaders. All operands are zero-padded in the workspace, so loads
// are unguarded: two 16B vector loads per fragment per lane.
// CDNA5 wave32 WMMA VGPR layouts (ISA 7.12.2):
//   A 16x32 f16 : lane L, half h -> m = L%16, k = 16*(h/8) + 8*(L/16) + h%8
//   B 32x16 f16 : lane L, half h -> n = L%16, k as A (loaders use k = 16*(L/16) + h for both A and B)
//   C/D 16x16 f32: lane L, reg r -> m = r + 8*(L/16), n = L%16
// ---------------------------------------------------------------------------
__device__ __forceinline__ v16h fragA_at(const _Float16* p) {
  v8h lo = *(const v8h*)(p);        // k = kb .. kb+7
  v8h hi = *(const v8h*)(p + 8);    // k = kb+8 .. kb+15
  v16h a;
#pragma unroll
  for (int j = 0; j < 8; ++j) { a[j] = lo[j]; a[8 + j] = hi[j]; }
  return a;
}

__device__ __forceinline__ v16h fragB_at(const _Float16* p) {
  v8h lo = *(const v8h*)(p);        // k = kb .. kb+7
  v8h hi = *(const v8h*)(p + 8);    // k = kb+8 .. kb+15
  v16h b;
#pragma unroll
  for (int j = 0; j < 8; ++j) { b[j] = lo[j]; b[8 + j] = hi[j]; }
  return b;
}

__device__ __forceinline__ const _Float16* laneA_base(const _Float16* mat, int ld,
                                                      int r0, int lane) {
  return mat + (size_t)(r0 + (lane & 15)) * ld + ((lane >> 4) << 4);
}
__device__ __forceinline__ const _Float16* laneB_base(const _Float16* mat, int ld,
                                                      int n0, int lane) {
  return mat + (size_t)(n0 + (lane & 15)) * ld + ((lane >> 4) << 4);
}

__device__ __forceinline__ v16h load_frag_A(const _Float16* mat, int ld, int r0,
                                            int k0, int lane) {
  return fragA_at(laneA_base(mat, ld, r0, lane) + k0);
}
__device__ __forceinline__ v16h load_frag_B(const _Float16* mat, int ld, int n0,
                                            int k0, int lane) {
  return fragB_at(laneB_base(mat, ld, n0, lane) + k0);
}

// ---------------------------------------------------------------------------
// Kernel 1: transpose features -> padded token-major f16 x-matrices (zero pad
// rows), convert W to f16.
//   xa: [5][128][640]   xb: [300][32][640]   w16: [384][640]
// ---------------------------------------------------------------------------
__global__ void cvt_kernel(const float* __restrict__ fa, const float* __restrict__ fb,
                           const float* __restrict__ W,
                           _Float16* __restrict__ xa, _Float16* __restrict__ xb,
                           _Float16* __restrict__ w16) {
  const int NA = NWAY * LA_P * CDIM;     // 409600
  const int NB = NQRY * LB_P * CDIM;     // 6144000
  const int NWEL = EDIM * CDIM;          // 245760
  int idx = blockIdx.x * 256 + threadIdx.x;
  if (idx < NA) {
    int t = idx / CDIM, d = idx - t * CDIM;
    int w = t >> 7, l = t & (LA_P - 1);
    xa[idx] = (l < LA) ? (_Float16)fa[((size_t)w * CDIM + d) * LA + l]
                       : (_Float16)0.0f;
  } else if (idx < NA + NB) {
    int i = idx - NA;
    int t = i / CDIM, d = i - t * CDIM;
    int q = t >> 5, l = t & (LB_P - 1);
    xb[i] = (l < LB) ? (_Float16)fb[((size_t)q * CDIM + d) * LB + l]
                     : (_Float16)0.0f;
  } else if (idx < NA + NB + NWEL) {
    int i = idx - NA - NB;
    w16[i] = (_Float16)W[i];
  }
}

// ---------------------------------------------------------------------------
// Kernel 2: QKV projection GEMM via WMMA.
//   out[t,e] = sum_d x[t,d] * W[e,d]
// Blocks: 40 for batch A (5 x 8 row-tiles), 600 for batch B (300 x 2).
// 4 waves/block, wave handles 6 of the 24 e-tiles.
// K loop fully unrolled: all loads use immediate offsets from a fixed per-lane
// pointer, so regalloc can feed WMMA operands directly (no copies) and the
// scheduler hoists loads ahead of the serial acc chain.
// ---------------------------------------------------------------------------
__global__ __launch_bounds__(128) void qkv_kernel(
    const _Float16* __restrict__ xa, const _Float16* __restrict__ xb,
    const _Float16* __restrict__ w16,
    _Float16* __restrict__ qa, _Float16* __restrict__ ka,
    _Float16* __restrict__ vaT, float* __restrict__ va32,
    _Float16* __restrict__ qb, _Float16* __restrict__ kb,
    _Float16* __restrict__ vbT, float* __restrict__ vb32) {
  const int lane = threadIdx.x & 31;
  const int wave = threadIdx.x >> 5;
  const int bidx = blockIdx.x;
  const bool isA = (bidx < NWAY * (LA_P / 16));
  int b, mtile;
  const _Float16* x;
  if (isA) {
    b = bidx >> 3; mtile = bidx & 7;
    x = xa + (size_t)b * LA_P * CDIM;
  } else {
    int t = bidx - NWAY * (LA_P / 16);
    b = t >> 1; mtile = t & 1;
    x = xb + (size_t)b * LB_P * CDIM;
  }
  const _Float16* pA = laneA_base(x, CDIM, mtile * 16, lane);

#pragma unroll 1
  for (int j = 0; j < 6; ++j) {
    const int et = wave + 4 * j;          // e-tile 0..23
    const _Float16* pB = laneB_base(w16, CDIM, et * 16, lane);

    v8f acc = {};
#pragma unroll
    for (int kt = 0; kt < CDIM / 32; ++kt) {   // 20 WMMAs, straight-line
      v16h A = fragA_at(pA + kt * 32);
      v16h B = fragB_at(pB + kt * 32);
      acc = WMMA_F16(A, B, acc);
    }

    const int m0 = mtile * 16 + ((lane >> 4) << 3);
    const int n0 = et * 16 + (lane & 15);   // e index 0..383
    const int d  = n0 & (DMODEL - 1);
#pragma unroll
    for (int r = 0; r < 8; ++r) {
      const int m = m0 + r;
      const float val = acc[r];
      if (isA) {
        const size_t base = (size_t)b * LA_P * DMODEL;
        if (n0 < DMODEL)          qa[base + (size_t)m * DMODEL + d] = (_Float16)val;
        else if (n0 < 2 * DMODEL) ka[base + (size_t)m * DMODEL + d] = (_Float16)val;
        else {
          vaT[base + (size_t)d * LA_P + m]    = (_Float16)val;   // [d][m]
          va32[base + (size_t)m * DMODEL + d] = val;
        }
      } else {
        const size_t base = (size_t)b * LB_P * DMODEL;
        if (n0 < DMODEL)          qb[base + (size_t)m * DMODEL + d] = (_Float16)val;
        else if (n0 < 2 * DMODEL) kb[base + (size_t)m * DMODEL + d] = (_Float16)val;
        else {
          vbT[(size_t)b * DMODEL * LB_P + (size_t)d * LB_P + m] = (_Float16)val;
          vb32[base + (size_t)m * DMODEL + d] = val;
        }
      }
    }
  }
}

// ---------------------------------------------------------------------------
// Kernel 3: one block per (q,w) pair. 4 waves / 128 threads.
//   BA: S = qb @ ka^T (25x125), softmax over 125, recon_b = P @ v_a, qs diff.
//   AB: S = qa @ kb^T (125x25), softmax over 25,  recon_a = P @ v_b, sq diff.
// All padding is zero, so diff loops need no guards: pad rows give (0-0)^2.
// ---------------------------------------------------------------------------
__global__ __launch_bounds__(128) void attn_kernel(
    const _Float16* __restrict__ qa16, const _Float16* __restrict__ ka16,
    const _Float16* __restrict__ vaT16, const float* __restrict__ va32,
    const _Float16* __restrict__ qb16, const _Float16* __restrict__ kb16,
    const _Float16* __restrict__ vbT16, const float* __restrict__ vb32,
    float* __restrict__ out) {
  const float SCALE = 0.08838834764831845f;   // 1/sqrt(128)
  const int q = blockIdx.x / NWAY;
  const int w = blockIdx.x - q * NWAY;
  const int tid = threadIdx.x, lane = tid & 31, wave = tid >> 5;

  __shared__ __align__(16) float    S[LA_P * LB_P];    // 4096 f32 = 16KB
  __shared__ __align__(16) _Float16 P[LA_P * LB_P];    // 4096 f16 = 8KB
  __shared__ float red[128];
  __shared__ float totals[1];

  const _Float16* qb  = qb16  + (size_t)q * LB_P * DMODEL;
  const _Float16* ka  = ka16  + (size_t)w * LA_P * DMODEL;
  const _Float16* vaT = vaT16 + (size_t)w * DMODEL * LA_P;
  const float*    vb  = vb32  + (size_t)q * LB_P * DMODEL;
  const _Float16* qa  = qa16  + (size_t)w * LA_P * DMODEL;
  const _Float16* kb  = kb16  + (size_t)q * LB_P * DMODEL;
  const _Float16* vbT = vbT16 + (size_t)q * DMODEL * LB_P;
  const float*    va  = va32  + (size_t)w * LA_P * DMODEL;

  float qs_local = 0.0f, sq_local = 0.0f;

  // ---- BA scores: S[32][128] = qb @ ka^T ----
  for (int i = 0; i < 4; ++i) {
    const int t = wave * 4 + i, mt = t >> 3, nt = t & 7;
    v8f acc = {};
#pragma unroll
    for (int kt = 0; kt < 4; ++kt) {
      v16h A = load_frag_A(qb, DMODEL, mt * 16, kt * 32, lane);
      v16h B = load_frag_B(ka, DMODEL, nt * 16, kt * 32, lane);
      acc = WMMA_F16(A, B, acc);
    }
    const int m0 = mt * 16 + ((lane >> 4) << 3), n = nt * 16 + (lane & 15);
#pragma unroll
    for (int r = 0; r < 8; ++r) S[(m0 + r) * LA_P + n] = acc[r] * SCALE;
  }
  __syncthreads();

  // ---- BA softmax over 125 valid cols, parallel: 32 rows x 4 col-slices ----
  {
    const int row = tid >> 2, sub = tid & 3;
    const int c0 = sub * 32;
    const int cend = (c0 + 32 < LA) ? (c0 + 32) : LA;   // 32,64,96,125
    float mx = -1e30f;
    for (int c = c0; c < cend; ++c) mx = fmaxf(mx, S[row * LA_P + c]);
    red[tid] = mx;
    __syncthreads();
    const float rmx = fmaxf(fmaxf(red[row * 4 + 0], red[row * 4 + 1]),
                            fmaxf(red[row * 4 + 2], red[row * 4 + 3]));
    float sum = 0.0f;
    for (int c = c0; c < cend; ++c) {
      float e = expf(S[row * LA_P + c] - rmx);
      S[row * LA_P + c] = e;
      sum += e;
    }
    __syncthreads();            // all reads of red done before overwrite
    red[tid] = sum;
    __syncthreads();
    const float inv = 1.0f / (red[row * 4 + 0] + red[row * 4 + 1] +
                              red[row * 4 + 2] + red[row * 4 + 3]);
    if (row < LB) {
      for (int c = c0; c < cend; ++c) P[row * LA_P + c] = (_Float16)(S[row * LA_P + c] * inv);
      for (int c = cend; c < c0 + 32; ++c) P[row * LA_P + c] = (_Float16)0.0f;
    } else {
      for (int c = c0; c < c0 + 32; ++c) P[row * LA_P + c] = (_Float16)0.0f;
    }
  }
  __syncthreads();

  // ---- recon_b = P @ v_a, fused qs diff ----
  for (int i = 0; i < 4; ++i) {
    const int t = wave * 4 + i, mt = t >> 3, nt = t & 7;
    v8f acc = {};
#pragma unroll
    for (int kt = 0; kt < 4; ++kt) {
      v16h A = load_frag_A(P, LA_P, mt * 16, kt * 32, lane);     // from LDS
      v16h B = load_frag_B(vaT, LA_P, nt * 16, kt * 32, lane);   // [d][m]
      acc = WMMA_F16(A, B, acc);
    }
    const int m0 = mt * 16 + ((lane >> 4) << 3), n = nt * 16 + (lane & 15);
#pragma unroll
    for (int r = 0; r < 8; ++r) {
      const float dl = vb[(size_t)(m0 + r) * DMODEL + n] - acc[r];
      qs_local += dl * dl;
    }
  }
  __syncthreads();

  // ---- AB scores: S[128][32] = qa @ kb^T ----
  for (int i = 0; i < 4; ++i) {
    const int t = wave * 4 + i, mt = t >> 1, nt = t & 1;
    v8f acc = {};
#pragma unroll
    for (int kt = 0; kt < 4; ++kt) {
      v16h A = load_frag_A(qa, DMODEL, mt * 16, kt * 32, lane);
      v16h B = load_frag_B(kb, DMODEL, nt * 16, kt * 32, lane);
      acc = WMMA_F16(A, B, acc);
    }
    const int m0 = mt * 16 + ((lane >> 4) << 3), n = nt * 16 + (lane & 15);
#pragma unroll
    for (int r = 0; r < 8; ++r) S[(m0 + r) * LB_P + n] = acc[r] * SCALE;
  }
  __syncthreads();

  // ---- AB softmax over 25 valid cols (125 rows, one thread per row) ----
  {
    const int row = tid;
    if (row < LA) {
      float mx = -1e30f;
      for (int c = 0; c < LB; ++c) mx = fmaxf(mx, S[row * LB_P + c]);
      float sum = 0.0f;
      for (int c = 0; c < LB; ++c) { float e = expf(S[row * LB_P + c] - mx); S[row * LB_P + c] = e; sum += e; }
      const float inv = 1.0f / sum;
      for (int c = 0; c < LB; ++c) P[row * LB_P + c] = (_Float16)(S[row * LB_P + c] * inv);
      for (int c = LB; c < LB_P; ++c) P[row * LB_P + c] = (_Float16)0.0f;
    } else {
      for (int c = 0; c < LB_P; ++c) P[row * LB_P + c] = (_Float16)0.0f;
    }
  }
  __syncthreads();

  // ---- recon_a = P @ v_b, fused sq diff (K=32, one WMMA per tile) ----
  for (int i = 0; i < 16; ++i) {
    const int t = wave * 16 + i, mt = t >> 3, nt = t & 7;
    v16h A = load_frag_A(P, LB_P, mt * 16, 0, lane);     // from LDS
    v16h B = load_frag_B(vbT, LB_P, nt * 16, 0, lane);   // [d][m]
    v8f acc = {};
    acc = WMMA_F16(A, B, acc);
    const int m0 = mt * 16 + ((lane >> 4) << 3), n = nt * 16 + (lane & 15);
#pragma unroll
    for (int r = 0; r < 8; ++r) {
      const float dl = va[(size_t)(m0 + r) * DMODEL + n] - acc[r];
      sq_local += dl * dl;
    }
  }

  // ---- block reductions ----
  red[tid] = qs_local; __syncthreads();
  for (int s = 64; s > 0; s >>= 1) { if (tid < s) red[tid] += red[tid + s]; __syncthreads(); }
  if (tid == 0) totals[0] = red[0];
  __syncthreads();
  red[tid] = sq_local; __syncthreads();
  for (int s = 64; s > 0; s >>= 1) { if (tid < s) red[tid] += red[tid + s]; __syncthreads(); }
  if (tid == 0) {
    out[blockIdx.x] = -red[0];                       // sq_similarity [300][5]
    out[NQRY * NWAY + blockIdx.x] = -totals[0];      // qs_similarity [300][5]
  }
}

// ---------------------------------------------------------------------------
// Host-side launch (graph-capture safe: only kernel launches on `stream`)
// ---------------------------------------------------------------------------
extern "C" void kernel_launch(void* const* d_in, const int* in_sizes, int n_in,
                              void* d_out, int out_size, void* d_ws, size_t ws_size,
                              hipStream_t stream) {
  const float* fa = (const float*)d_in[0];   // [5,640,125]
  const float* fb = (const float*)d_in[1];   // [300,640,25]
  const float* W  = (const float*)d_in[2];   // [384,640]
  float* out = (float*)d_out;                // 3000 f32 (sq then qs)

  char* ws = (char*)d_ws;
  size_t off = 0;
  auto carve = [&](size_t bytes) -> void* {
    void* p = ws + off;
    off += (bytes + 255) & ~(size_t)255;
    return p;
  };
  _Float16* xa16 = (_Float16*)carve((size_t)NWAY * LA_P * CDIM * 2);
  _Float16* xb16 = (_Float16*)carve((size_t)NQRY * LB_P * CDIM * 2);
  _Float16* w16  = (_Float16*)carve((size_t)EDIM * CDIM * 2);
  _Float16* qa16 = (_Float16*)carve((size_t)NWAY * LA_P * DMODEL * 2);
  _Float16* ka16 = (_Float16*)carve((size_t)NWAY * LA_P * DMODEL * 2);
  _Float16* vaT  = (_Float16*)carve((size_t)NWAY * DMODEL * LA_P * 2);
  float*    va32 = (float*)   carve((size_t)NWAY * LA_P * DMODEL * 4);
  _Float16* qb16 = (_Float16*)carve((size_t)NQRY * LB_P * DMODEL * 2);
  _Float16* kb16 = (_Float16*)carve((size_t)NQRY * LB_P * DMODEL * 2);
  _Float16* vbT  = (_Float16*)carve((size_t)NQRY * DMODEL * LB_P * 2);
  float*    vb32 = (float*)   carve((size_t)NQRY * LB_P * DMODEL * 4);
  (void)ws_size; (void)in_sizes; (void)n_in; (void)out_size;

  const int cvt_total = NWAY * LA_P * CDIM + NQRY * LB_P * CDIM + EDIM * CDIM;
  cvt_kernel<<<(cvt_total + 255) / 256, 256, 0, stream>>>(fa, fb, W, xa16, xb16, w16);

  const int qkv_blocks = NWAY * (LA_P / 16) + NQRY * (LB_P / 16);  // 40 + 600
  qkv_kernel<<<qkv_blocks, 128, 0, stream>>>(xa16, xb16, w16,
                                             qa16, ka16, vaT, va32,
                                             qb16, kb16, vbT, vb32);

  attn_kernel<<<NQRY * NWAY, 128, 0, stream>>>(qa16, ka16, vaT, va32,
                                               qb16, kb16, vbT, vb32, out);
}
